// CKConv_10694468567662
// MI455X (gfx1250) — matrix operand order repaired and checked by reading, not compiled, not run on hardware
//
#include <hip/hip_runtime.h>

#define OMEGA0 30.0f
#define H 32
#define WAVES 8

typedef _Float16 v16h __attribute__((ext_vector_type(16)));
typedef _Float16 v8h  __attribute__((ext_vector_type(8)));
typedef float    v8f  __attribute__((ext_vector_type(8)));
typedef float    v4f  __attribute__((ext_vector_type(4)));

__device__ __forceinline__ void dswait() {
    __asm__ volatile("s_wait_dscnt 0" ::: "memory");
}
__device__ __forceinline__ void asyncwait() {
    __asm__ volatile("s_wait_asynccnt 0" ::: "memory");
}
// Async DMA: each lane copies 16B from global to LDS (tracked on ASYNCcnt).
__device__ __forceinline__ void async_load_b128(unsigned lds_off, const void* gaddr) {
    __asm__ volatile("global_load_async_to_lds_b128 %0, %1, off"
                     :: "v"(lds_off), "v"(gaddr) : "memory");
}

__global__ void zero_kernel(float* __restrict__ out, int n) {
    int i = blockIdx.x * blockDim.x + threadIdx.x;
    if (i < n) out[i] = 0.0f;
}

// LDS layout (dynamic):
//   [0, 65536)          : W3 as f16, raw order w3[k*1024 + h*32 + d]   (32768 halves)
//   [65536, 69632)      : W2 f32 (1024)
//   [69632, 69760)      : W1 f32 (32)
//   [69760 + wid*3392)  : per-wave scratch:
//       +0    t[16] f32
//       +64   seg[16] i32
//       +128  gat[16] i32
//       +192  h1[32] f32
//       +320  h2 f16 [16][32]            (1024 B)
//       +1344 emb f32 staging [16][32]   (2048 B, async-DMA target)
#define SMEM_W2   65536
#define SMEM_W1   (65536 + 4096)
#define SMEM_WAVE 69760
#define WAVE_SZ   3392
#define SMEM_TOTAL (SMEM_WAVE + WAVES * WAVE_SZ)

__launch_bounds__(256)
__global__ void siren_msg_kernel(
    const float* __restrict__ u_emb, const float* __restrict__ i_emb,
    const int*   __restrict__ user_idx, const int* __restrict__ item_idx,
    const float* __restrict__ edges_t,
    const float* __restrict__ u_t, const float* __restrict__ i_t,
    const float* __restrict__ Wu1, const float* __restrict__ Wu2, const float* __restrict__ Wu3,
    const float* __restrict__ Wi1, const float* __restrict__ Wi2, const float* __restrict__ Wi3,
    float* __restrict__ out, int U, int I, int E)
{
    extern __shared__ char smem[];
    _Float16* w3h = (_Float16*)smem;
    float*    w2  = (float*)(smem + SMEM_W2);
    float*    w1  = (float*)(smem + SMEM_W1);

    const int wid = threadIdx.x >> 5;
    const int lid = threadIdx.x & 31;
    char* wbase = smem + SMEM_WAVE + wid * WAVE_SZ;
    float*    t_lds   = (float*)(wbase);
    int*      seg_lds = (int*)  (wbase + 64);
    int*      gat_lds = (int*)  (wbase + 128);
    float*    h1_lds  = (float*)(wbase + 192);
    _Float16* h2h     = (_Float16*)(wbase + 320);
    float*    embf    = (float*)(wbase + 1344);

    // side 0: item kernels (Wi, i_t, i_emb gathered by item) -> hLu[user]
    // side 1: user kernels (Wu, u_t, u_emb gathered by user) -> hLi[item]
    const int side = blockIdx.y;
    const float* W1 = side ? Wu1 : Wi1;
    const float* W2 = side ? Wu2 : Wi2;
    const float* W3 = side ? Wu3 : Wi3;
    const float* tt = side ? u_t : i_t;
    const int* gidx = side ? user_idx : item_idx;   // gather index
    const int* sidx = side ? item_idx : user_idx;   // segment (scatter) index
    const float* emb = side ? u_emb : i_emb;
    float* obase = out + (side ? (size_t)U * H : 0);

    // ---- stage weights into LDS (block-cooperative) ----
    for (int i = threadIdx.x; i < H * H * H; i += blockDim.x) w3h[i] = (_Float16)W3[i];
    for (int i = threadIdx.x; i < H * H; i += blockDim.x)     w2[i]  = W2[i];
    if (threadIdx.x < H)                                      w1[threadIdx.x] = W1[threadIdx.x];
    __syncthreads();

    const int tile = blockIdx.x * WAVES + wid;      // 16 edges per wave
    if (tile * 16 >= E) return;

    const int row = lid & 15;
    const int hi  = lid >> 4;
    const int n   = lid & 15;

    // ---- per-edge scalars: t, gather idx, segment idx ----
    if (lid < 16) {
        int e = tile * 16 + lid;
        if (e < E) {
            int g = gidx[e];
            gat_lds[lid] = g;
            seg_lds[lid] = sidx[e];
            t_lds[lid]   = tt[g] - edges_t[e];
        } else {
            gat_lds[lid] = 0; seg_lds[lid] = 0; t_lds[lid] = 0.0f;  // t=0 -> h2=0 -> msg=0
        }
    }
    dswait();

    // ---- kick off async DMA of gathered embedding rows into LDS (f32) ----
    // 4 ops x 32 lanes x 16B = 2KB (16 rows x 128B); overlapped with SIREN math below.
    {
        const int cq = lid & 7;          // 16B chunk within a 128B row
        #pragma unroll
        for (int q = 0; q < 4; ++q) {
            int r = q * 4 + (lid >> 3);  // row handled by this lane for op q
            int g = gat_lds[r];
            unsigned dst = (unsigned)(uintptr_t)(embf + r * H + cq * 4);
            async_load_b128(dst, emb + (size_t)g * H + cq * 4);
        }
    }

    // ---- SIREN layers 1 & 2 (lane = feature j) ----
    for (int m = 0; m < 16; ++m) {
        float tm = t_lds[m];
        h1_lds[lid] = __sinf(OMEGA0 * tm * w1[lid]);
        dswait();
        float acc = 0.0f;
        #pragma unroll
        for (int k = 0; k < H; ++k) acc += h1_lds[k] * w2[k * H + lid];
        h2h[m * H + lid] = (_Float16)__sinf(OMEGA0 * acc);
        dswait();
    }

    // ---- build A-side emb fragment from LDS staging (async DMA results) ----
    asyncwait();
    // A frag (16-bit 16x32): lane holds row=lid&15; halves 0..7 = K=hi*8+j, 8..15 = K=16+hi*8+j
    v4f f0 = *(const v4f*)(embf + row * H + hi * 8);
    v4f f1 = *(const v4f*)(embf + row * H + hi * 8 + 4);
    v4f f2 = *(const v4f*)(embf + row * H + 16 + hi * 8);
    v4f f3 = *(const v4f*)(embf + row * H + 16 + hi * 8 + 4);
    v16h ea;
    #pragma unroll
    for (int i = 0; i < 4; ++i) {
        ea[i]      = (_Float16)f0[i];
        ea[4 + i]  = (_Float16)f1[i];
        ea[8 + i]  = (_Float16)f2[i];
        ea[12 + i] = (_Float16)f3[i];
    }

    // ---- GEMM: [16 x 1024] (outer(h2,emb)) @ [1024 x 32] (W3') via WMMA ----
    v8f c0 = {};   // h = 0..15
    v8f c1 = {};   // h = 16..31
    #pragma unroll 4
    for (int s = 0; s < 32; ++s) {          // K-step s: kd = s*32 + d
        _Float16 hv = h2h[row * H + s];
        v16h a = ea * hv;                   // A[row, d] = h2[row,s] * emb[row,d]

        // B frag (32x16): lane col = n (+ ntile*16), halves 0..7 = K=hi*8+j, 8..15 = K=16+hi*8+j
        const _Float16* bp0 = w3h + s * 1024 + n * H + hi * 8;
        const _Float16* bp1 = w3h + s * 1024 + (16 + n) * H + hi * 8;
        v8h b0a = *(const v8h*)(bp0);
        v8h b0b = *(const v8h*)(bp0 + 16);
        v8h b1a = *(const v8h*)(bp1);
        v8h b1b = *(const v8h*)(bp1 + 16);
        v16h b0 = __builtin_shufflevector(b0a, b0b, 0,1,2,3,4,5,6,7,8,9,10,11,12,13,14,15);
        v16h b1 = __builtin_shufflevector(b1a, b1b, 0,1,2,3,4,5,6,7,8,9,10,11,12,13,14,15);

        c0 = __builtin_amdgcn_wmma_f32_16x16x32_f16(false, a, false, b0, (short)0, c0, false, false);
        c1 = __builtin_amdgcn_wmma_f32_16x16x32_f16(false, a, false, b1, (short)0, c1, false, false);
    }

    // ---- scatter-add messages: D layout VGPR r -> M = hi*8 + r, N = n ----
    #pragma unroll
    for (int r = 0; r < 8; ++r) {
        int M = hi * 8 + r;
        int e = tile * 16 + M;
        if (e < E) {
            int seg = seg_lds[M];
            float* dst = obase + (size_t)seg * H + n;
            unsafeAtomicAdd(dst,      c0[r]);   // -> global_atomic_add_f32
            unsafeAtomicAdd(dst + 16, c1[r]);
        }
    }
}

extern "C" void kernel_launch(void* const* d_in, const int* in_sizes, int n_in,
                              void* d_out, int out_size, void* d_ws, size_t ws_size,
                              hipStream_t stream) {
    const float* u_emb    = (const float*)d_in[0];
    const float* i_emb    = (const float*)d_in[1];
    const int*   user_idx = (const int*)  d_in[2];
    const int*   item_idx = (const int*)  d_in[3];
    const float* edges_t  = (const float*)d_in[4];
    const float* u_t      = (const float*)d_in[5];
    const float* i_t      = (const float*)d_in[6];
    const float* Wu1      = (const float*)d_in[7];
    const float* Wu2      = (const float*)d_in[8];
    const float* Wu3      = (const float*)d_in[9];
    const float* Wi1      = (const float*)d_in[10];
    const float* Wi2      = (const float*)d_in[11];
    const float* Wi3      = (const float*)d_in[12];

    int E = in_sizes[4];
    int U = in_sizes[5];
    int I = in_sizes[6];
    float* out = (float*)d_out;

    // outputs are accumulated with atomics; zero the poisoned buffer first
    zero_kernel<<<(out_size + 255) / 256, 256, 0, stream>>>(out, out_size);

    int tiles = (E + 15) / 16;
    dim3 grid((tiles + WAVES - 1) / WAVES, 2);
    // allow >64KB dynamic LDS (idempotent; ignore failure on toolchains that don't need it)
    (void)hipFuncSetAttribute((const void*)siren_msg_kernel,
                              hipFuncAttributeMaxDynamicSharedMemorySize, SMEM_TOTAL);
    siren_msg_kernel<<<grid, 256, SMEM_TOTAL, stream>>>(
        u_emb, i_emb, user_idx, item_idx, edges_t, u_t, i_t,
        Wu1, Wu2, Wu3, Wi1, Wi2, Wi3, out, U, I, E);
}
